// MouseSRNN_74036646248787
// MI455X (gfx1250) — matrix operand order repaired and checked by reading, not compiled, not run on hardware
//
#include <hip/hip_runtime.h>
#include <stdint.h>
#include <math.h>

#define NTHREADS 256
#define NWAVES 8

typedef __attribute__((ext_vector_type(16))) _Float16 v16h;
typedef __attribute__((ext_vector_type(8)))  float    v8f;
typedef __attribute__((ext_vector_type(4)))  float    f32x4;
typedef __attribute__((ext_vector_type(4)))  unsigned u32x4;

union HFrag { v16h v; _Float16 h[16]; unsigned u[8]; u32x4 q[2]; };

// ---------------- LDS layout (float element offsets) ----------------
#define O_HT   0        // 16x128
#define O_CT   2048     // 16x128
#define O_HN   4096     // 16x128
#define O_CN   6144     // 16x128
#define O_HS   8192     // 144x128
#define O_S0   26624    // 144x64  (se act, later Pi)
#define O_S1   35840    // 144x64  (Pe)
#define O_FEAT 45056    // 144x32
#define O_TE   49664    // 16x64
#define O_QB   50688    // 16x64
#define O_CAT  51712    // 16x384
#define O_NI   57856    // 16x128
#define O_SCI  59904    // 12x3
#define O_SCE  59940    // 12x8
#define O_WI   60036    // 12x3
#define O_WE   60072    // 12x8
#define O_VI   60168    // 64
#define O_VE   60232    // 64
#define O_KP   60296    // 32
#define O_RED  60328    // 12
#define SM_FLOATS 60352
// int table layout
#define I_INTRA 0       // 36
#define I_INTER 36      // 96
#define I_SRC   132     // 132
#define I_DST   264     // 132

// ---------------- explicit address-space helpers ----------------
__device__ __forceinline__ float gloadf(const float* p) {
  return *(const __attribute__((address_space(1))) float*)(uintptr_t)p;
}
__device__ __forceinline__ int gloadi(const int* p) {
  return *(const __attribute__((address_space(1))) int*)(uintptr_t)p;
}
__device__ __forceinline__ u32x4 gloadq(const unsigned* p) {
  return *(const __attribute__((address_space(1))) u32x4*)(uintptr_t)p;
}
__device__ __forceinline__ void gstoref(float* p, float v) {
  *(__attribute__((address_space(1))) float*)(uintptr_t)p = v;
}
__device__ __forceinline__ void gstoreu(unsigned* p, unsigned v) {
  *(__attribute__((address_space(1))) unsigned*)(uintptr_t)p = v;
}

__device__ __forceinline__ v8f wmma16(v16h a, v16h b, v8f c) {
  return __builtin_amdgcn_wmma_f32_16x16x32_f16(false, a, false, b, (short)0, c, false, false);
}

__device__ __forceinline__ float sigmoidf_(float x) { return 1.f / (1.f + __expf(-x)); }

// A fragment from the shared buffer.  16-bit A 16x32 layout: lane<16 -> Kbase 0,
// lane>=16 -> Kbase 8; a lane needs row[kb+0..7] and row[kb+16..23]:
// four 16-byte LDS vector loads (all offsets 16B aligned by layout construction).
__device__ __forceinline__ v16h load_a_frag(const float* sm, int base, int ld, int kofs,
                                            int mt, int lane) {
  int m   = mt * 16 + (lane & 15);
  int kb  = (lane < 16) ? 0 : 8;
  int row = base + m * ld + kofs + kb;
  f32x4 a0 = *(const f32x4*)(sm + row);
  f32x4 a1 = *(const f32x4*)(sm + row + 4);
  f32x4 a2 = *(const f32x4*)(sm + row + 16);
  f32x4 a3 = *(const f32x4*)(sm + row + 20);
  HFrag f;
#pragma unroll
  for (int i = 0; i < 4; ++i) {
    f.h[i]      = (_Float16)a0[i];
    f.h[4 + i]  = (_Float16)a1[i];
    f.h[8 + i]  = (_Float16)a2[i];
    f.h[12 + i] = (_Float16)a3[i];
  }
  return f.v;
}

// B fragment from pre-packed global weights: [ks][nt][lane][j] u32 -> 2 x b128 per lane
__device__ __forceinline__ v16h load_b_frag(const unsigned* Wp, int ntiles, int ks, int nt, int lane) {
  const unsigned* p = Wp + (((size_t)ks * ntiles + nt) * 32 + lane) * 8;
  HFrag f;
  f.q[0] = gloadq(p);
  f.q[1] = gloadq(p + 4);
  return f.v;
}

// OUT tile (in LDS at ooff, stride ldo) = act(X @ W + bias)
template <int RELU>
__device__ void gemm_lds(float* sm, int xoff, int ldx, const unsigned* Wp, const float* bias,
                         int ooff, int ldo, int Mtiles, int Ntiles, int Ksteps,
                         int wave, int lane) {
  int tot = Mtiles * Ntiles;
  for (int j = wave; j < tot; j += NWAVES) {
    int mt = j / Ntiles, nt = j - mt * Ntiles;
    v8f acc = {0.f, 0.f, 0.f, 0.f, 0.f, 0.f, 0.f, 0.f};
    for (int ks = 0; ks < Ksteps; ++ks) {
      v16h a = load_a_frag(sm, xoff, ldx, ks * 32, mt, lane);
      v16h b = load_b_frag(Wp, Ntiles, ks, nt, lane);
      acc = wmma16(a, b, acc);
    }
    int n  = lane & 15;
    int mb = (lane < 16) ? 0 : 8;
    float bb = gloadf(bias + nt * 16 + n);
#pragma unroll
    for (int r = 0; r < 8; ++r) {
      float v = acc[r] + bb;
      if (RELU) v = fmaxf(v, 0.f);
      sm[ooff + (mt * 16 + r + mb) * ldo + nt * 16 + n] = v;
    }
  }
  __syncthreads();
}

// Fused LSTM step: gates = [X|H] @ Wpacked + bias (N=512, hidden=128), then c,h update.
// CG=true: cell state in global memory (spatial LSTM), else in LDS at coff.
// New h staged in registers; written back to LDS only after a barrier.
template <int JPW, bool CG>
__device__ void lstm_layer(float* sm, int xoff, int ldx, int kinSteps, int hoff,
                           float* Cg, int coff,
                           const unsigned* Wp, const float* bias,
                           int Mtiles, int Mrows, int wave, int lane) {
  const int tot    = Mtiles * 8;
  const int Ksteps = kinSteps + 4;
  float hst[JPW][8];
#pragma unroll
  for (int z = 0; z < JPW; ++z) {
    int j = wave + z * NWAVES;
    if (j < tot) {
      int mt = j >> 3, hq = j & 7;
      v8f aI = {0.f,0.f,0.f,0.f,0.f,0.f,0.f,0.f};
      v8f aF = aI, aG = aI, aO = aI;
      for (int ks = 0; ks < Ksteps; ++ks) {
        int base = (ks < kinSteps) ? xoff : hoff;
        int ld   = (ks < kinSteps) ? ldx  : 128;
        int kofs = (ks < kinSteps) ? ks * 32 : (ks - kinSteps) * 32;
        v16h a  = load_a_frag(sm, base, ld, kofs, mt, lane);
        v16h b0 = load_b_frag(Wp, 32, ks, hq,      lane);
        v16h b1 = load_b_frag(Wp, 32, ks, hq + 8,  lane);
        v16h b2 = load_b_frag(Wp, 32, ks, hq + 16, lane);
        v16h b3 = load_b_frag(Wp, 32, ks, hq + 24, lane);
        aI = wmma16(a, b0, aI);
        aF = wmma16(a, b1, aF);
        aG = wmma16(a, b2, aG);
        aO = wmma16(a, b3, aO);
      }
      int n = lane & 15;
      int mb = (lane < 16) ? 0 : 8;
      int hcol = hq * 16 + n;
      float bi = gloadf(bias + hcol), bf = gloadf(bias + 128 + hcol);
      float bg = gloadf(bias + 256 + hcol), bo = gloadf(bias + 384 + hcol);
#pragma unroll
      for (int r = 0; r < 8; ++r) {
        int m = mt * 16 + r + mb;
        float hv = 0.f;
        if (m < Mrows) {
          float gi = aI[r] + bi, gf = aF[r] + bf, gg = aG[r] + bg, go = aO[r] + bo;
          float cold = CG ? gloadf(Cg + (size_t)m * 128 + hcol) : sm[coff + m * 128 + hcol];
          float cnew = sigmoidf_(gf) * cold + sigmoidf_(gi) * tanhf(gg);
          if (CG) gstoref(Cg + (size_t)m * 128 + hcol, cnew);
          else    sm[coff + m * 128 + hcol] = cnew;
          hv = sigmoidf_(go) * tanhf(cnew);
        }
        hst[z][r] = hv;
      }
    } else {
#pragma unroll
      for (int r = 0; r < 8; ++r) hst[z][r] = 0.f;
    }
  }
  __syncthreads();
#pragma unroll
  for (int z = 0; z < JPW; ++z) {
    int j = wave + z * NWAVES;
    if (j < tot) {
      int mt = j >> 3, hq = j & 7;
      int n = lane & 15, mb = (lane < 16) ? 0 : 8;
#pragma unroll
      for (int r = 0; r < 8; ++r) {
        int m = mt * 16 + r + mb;
        if (m < Mrows) sm[hoff + m * 128 + hq * 16 + n] = hst[z][r];
      }
    }
  }
  __syncthreads();
}

// ---- weight packing: logical [Ktot][N] f32 (Wi rows [0,Kin), Wh rows [Kin,Kreal), zeros beyond)
// ---- -> f16 pairs laid out [ks][nt][lane][j] u32 matching the B-fragment layout.
__global__ void pack_kernel(const float* Wi, const float* Wh, int Kin, int Kreal,
                            int Ktot, int N, unsigned* dst) {
  int idx = blockIdx.x * blockDim.x + threadIdx.x;
  int ntiles = N >> 4;
  int total = (Ktot >> 5) * ntiles * 256;
  if (idx >= total) return;
  int j = idx & 7, lane = (idx >> 3) & 31;
  int rest = idx >> 8;
  int nt = rest % ntiles, ks = rest / ntiles;
  int col = nt * 16 + (lane & 15);
  int k0 = ks * 32 + ((lane < 16) ? 0 : 8) + 2 * j + ((j >= 4) ? 8 : 0);
  auto fetch = [&](int k) -> float {
    if (k < Kin)   return gloadf(Wi + (size_t)k * N + col);
    if (k < Kreal) return gloadf(Wh + (size_t)(k - Kin) * N + col);
    return 0.f;
  };
  HFrag f;
  f.h[0] = (_Float16)fetch(k0);
  f.h[1] = (_Float16)fetch(k0 + 1);
  gstoreu(dst + idx, f.u[0]);
}

struct P {
  const float *nodes, *et, *es;
  const float *kp_emb, *W_te, *b_te, *b_t, *b_se, *b_s;
  const float *b_q, *b_ki, *b_ke, *v_i, *v_e;
  const float *W_ne, *b_ne, *b_ea, *b_n, *W_out, *b_out;
  const int *src, *dst, *intra, *inter;
  const unsigned *WT, *WS, *WN, *WQ, *WKI, *WKE, *WEA, *WSE;
  float *cs;    // [B][144][128] spatial LSTM cell state (global scratch)
  float *entp;  // [B] entropy partials
  float *out;   // [B][T][12][5] (+1 scalar written by fin_kernel)
};

__global__ __launch_bounds__(NTHREADS, 1) void srnn_kernel(P p) {
  __shared__ float sm[SM_FLOATS];
  __shared__ int   smi[396];

  const int tid = threadIdx.x;
  const int wave = tid >> 5, lane = tid & 31;
  const int b = blockIdx.x;
  float* csb = p.cs + (size_t)b * 144 * 128;

  // ---- zero init (pad rows only need to be finite)
  for (int i = tid; i < SM_FLOATS; i += NTHREADS) sm[i] = 0.f;
  for (int i = tid; i < 144 * 128; i += NTHREADS) gstoref(csb + i, 0.f);
  if (tid < 36)  smi[I_INTRA + tid] = gloadi(p.intra + tid);
  if (tid < 96)  smi[I_INTER + tid] = gloadi(p.inter + tid);
  if (tid < 132) { smi[I_SRC + tid] = gloadi(p.src + tid); smi[I_DST + tid] = gloadi(p.dst + tid); }
  if (tid < 64)  { sm[O_VI + tid] = gloadf(p.v_i + tid); sm[O_VE + tid] = gloadf(p.v_e + tid); }
  if (tid < 32)  sm[O_KP + tid] = gloadf(p.kp_emb + tid);
  __syncthreads();

  float eloc = 0.f;

  for (int t = 0; t < 64; ++t) {
    // ---- temporal edge embedding: te = relu(et @ W_te + b_te)   (12x2 @ 2x64)
    const float* etv = p.et + ((size_t)b * 64 + t) * 24;
    for (int i = tid; i < 12 * 64; i += NTHREADS) {
      int n = i >> 6, c = i & 63;
      float v = gloadf(etv + n * 2) * gloadf(p.W_te + c)
              + gloadf(etv + n * 2 + 1) * gloadf(p.W_te + 64 + c) + gloadf(p.b_te + c);
      sm[O_TE + n * 64 + c] = fmaxf(v, 0.f);
    }
    __syncthreads();
    // ---- temporal LSTM (12 rows, K=64+128)
    lstm_layer<1, false>(sm, O_TE, 64, 2, O_HT, nullptr, O_CT, p.WT, p.b_t, 1, 12, wave, lane);

    // ---- spatial edge features (132 x 19, K padded to 32)
    const float* esv = p.es + ((size_t)b * 64 + t) * 264;
    for (int e = tid; e < 132; e += NTHREADS) {
      float x = gloadf(esv + e * 2), y = gloadf(esv + e * 2 + 1);
      float d = fmaxf(sqrtf(x * x + y * y), 1e-6f);
      int fb = O_FEAT + e * 32;
      sm[fb + 0] = x / d; sm[fb + 1] = y / d; sm[fb + 2] = logf(d);
      int sk = smi[I_SRC + e] * 8, dk = smi[I_DST + e] * 8;
#pragma unroll
      for (int i2 = 0; i2 < 8; ++i2) {
        sm[fb + 3 + i2]  = sm[O_KP + sk + i2];
        sm[fb + 11 + i2] = sm[O_KP + dk + i2];
      }
    }
    __syncthreads();
    // ---- se = relu(feat @ W_se + b_se) into S0 (act)
    gemm_lds<1>(sm, O_FEAT, 32, p.WSE, p.b_se, O_S0, 64, 9, 4, 1, wave, lane);
    // ---- spatial LSTM (132 rows, K=64+128), cell state in global scratch
    lstm_layer<9, true>(sm, O_S0, 64, 2, O_HS, csb, 0, p.WS, p.b_s, 9, 132, wave, lane);

    // ---- attention projections
    gemm_lds<0>(sm, O_HT, 128, p.WQ,  p.b_q,  O_QB, 64, 1, 4, 4, wave, lane);
    gemm_lds<0>(sm, O_HS, 128, p.WKI, p.b_ki, O_S0, 64, 9, 4, 4, wave, lane); // Pi
    gemm_lds<0>(sm, O_HS, 128, p.WKE, p.b_ke, O_S1, 64, 9, 4, 4, wave, lane); // Pe

    // ---- attention scores
    if (tid < 36) {
      int n = tid / 3, k = tid - 3 * n; int e = smi[I_INTRA + n * 3 + k];
      int pi = O_S0 + e * 64, qn = O_QB + n * 64;
      float s = 0.f;
      for (int a = 0; a < 64; ++a) s += tanhf(sm[qn + a] + sm[pi + a]) * sm[O_VI + a];
      sm[O_SCI + n * 3 + k] = s;
    }
    if (tid >= 64 && tid < 160) {
      int pp = tid - 64; int n = pp >> 3, k = pp & 7; int e = smi[I_INTER + n * 8 + k];
      int pe = O_S1 + e * 64, qn = O_QB + n * 64;
      float s = 0.f;
      for (int a = 0; a < 64; ++a) s += tanhf(sm[qn + a] + sm[pe + a]) * sm[O_VE + a];
      sm[O_SCE + n * 8 + k] = s;
    }
    __syncthreads();
    // ---- softmax + entropy (per node)
    if (tid < 12) {
      int n = tid;
      float s0 = sm[O_SCI + n * 3], s1 = sm[O_SCI + n * 3 + 1], s2 = sm[O_SCI + n * 3 + 2];
      float mx = fmaxf(fmaxf(s0, s1), s2);
      float e0 = __expf(s0 - mx), e1 = __expf(s1 - mx), e2 = __expf(s2 - mx);
      float ss = e0 + e1 + e2;
      float w0 = e0 / ss, w1 = e1 / ss, w2 = e2 / ss;
      sm[O_WI + n * 3] = w0; sm[O_WI + n * 3 + 1] = w1; sm[O_WI + n * 3 + 2] = w2;
      eloc += w0 * logf(w0 + 1e-10f) + w1 * logf(w1 + 1e-10f) + w2 * logf(w2 + 1e-10f);
      float sv[8];
#pragma unroll
      for (int k = 0; k < 8; ++k) sv[k] = sm[O_SCE + n * 8 + k];
      float mx2 = sv[0];
#pragma unroll
      for (int k = 1; k < 8; ++k) mx2 = fmaxf(mx2, sv[k]);
      float sum = 0.f;
#pragma unroll
      for (int k = 0; k < 8; ++k) { sv[k] = __expf(sv[k] - mx2); sum += sv[k]; }
#pragma unroll
      for (int k = 0; k < 8; ++k) {
        float w = sv[k] / sum; sm[O_WE + n * 8 + k] = w;
        eloc += w * logf(w + 1e-10f);
      }
    }
    __syncthreads();
    // ---- cat = [h_t | h_intra | h_inter]  and node_in
    for (int i = tid; i < 12 * 128; i += NTHREADS) {
      int n = i >> 7, c = i & 127;
      sm[O_CAT + n * 384 + c] = sm[O_HT + n * 128 + c];
      float hi = 0.f, he = 0.f;
#pragma unroll
      for (int k = 0; k < 3; ++k) hi += sm[O_WI + n * 3 + k] * sm[O_HS + smi[I_INTRA + n * 3 + k] * 128 + c];
#pragma unroll
      for (int k = 0; k < 8; ++k) he += sm[O_WE + n * 8 + k] * sm[O_HS + smi[I_INTER + n * 8 + k] * 128 + c];
      sm[O_CAT + n * 384 + 128 + c] = hi;
      sm[O_CAT + n * 384 + 256 + c] = he;
    }
    const float* nod = p.nodes + ((size_t)b * 64 + t) * 24;
    for (int i = tid; i < 12 * 64; i += NTHREADS) {
      int n = i >> 6, c = i & 63;
      float v = gloadf(nod + n * 2) * gloadf(p.W_ne + c)
              + gloadf(nod + n * 2 + 1) * gloadf(p.W_ne + 64 + c) + gloadf(p.b_ne + c);
      sm[O_NI + n * 128 + c] = fmaxf(v, 0.f);
    }
    __syncthreads();
    // ---- edge_in = relu(cat @ W_ea + b_ea) into ni[:, 64:128)
    gemm_lds<1>(sm, O_CAT, 384, p.WEA, p.b_ea, O_NI + 64, 128, 1, 4, 12, wave, lane);
    // ---- node LSTM (12 rows, K=128+128)
    lstm_layer<1, false>(sm, O_NI, 128, 4, O_HN, nullptr, O_CN, p.WN, p.b_n, 1, 12, wave, lane);
    // ---- output projection (12 x 5)
    for (int i = tid; i < 60; i += NTHREADS) {
      int n = i / 5, jj = i - 5 * n;
      float s = gloadf(p.b_out + jj);
      for (int c = 0; c < 128; ++c) s += sm[O_HN + n * 128 + c] * gloadf(p.W_out + c * 5 + jj);
      gstoref(p.out + (((size_t)b * 64 + t) * 12 + n) * 5 + jj, s);
    }
    __syncthreads();
  }

  if (tid < 12) sm[O_RED + tid] = eloc;
  __syncthreads();
  if (tid == 0) {
    float s = 0.f;
    for (int i = 0; i < 12; ++i) s += sm[O_RED + i];
    gstoref(p.entp + b, s);
  }
}

__global__ void fin_kernel(const float* entp, float* out) {
  if (blockIdx.x == 0 && threadIdx.x == 0) {
    float s = 0.f;
#pragma unroll 1
    for (int i = 0; i < 128; ++i) s += gloadf(entp + i);
    gstoref(out + 491520, -s / 98304.f);  // /(B*N)/T = /(128*12*64)
  }
}

extern "C" void kernel_launch(void* const* d_in, const int* in_sizes, int n_in,
                              void* d_out, int out_size, void* d_ws, size_t ws_size,
                              hipStream_t stream) {
  (void)in_sizes; (void)n_in; (void)out_size; (void)ws_size;
  const float* nodes = (const float*)d_in[0];
  const float* et    = (const float*)d_in[1];
  const float* es    = (const float*)d_in[2];
  const float* kp_emb= (const float*)d_in[3];
  const float* W_te  = (const float*)d_in[4];
  const float* b_te  = (const float*)d_in[5];
  const float* Wi_t  = (const float*)d_in[6];
  const float* Wh_t  = (const float*)d_in[7];
  const float* b_t   = (const float*)d_in[8];
  const float* W_se  = (const float*)d_in[9];
  const float* b_se  = (const float*)d_in[10];
  const float* Wi_s  = (const float*)d_in[11];
  const float* Wh_s  = (const float*)d_in[12];
  const float* b_s   = (const float*)d_in[13];
  const float* W_q   = (const float*)d_in[14];
  const float* b_q   = (const float*)d_in[15];
  const float* W_ki  = (const float*)d_in[16];
  const float* b_ki  = (const float*)d_in[17];
  const float* W_ke  = (const float*)d_in[18];
  const float* b_ke  = (const float*)d_in[19];
  const float* v_i   = (const float*)d_in[20];
  const float* v_e   = (const float*)d_in[22];
  const float* W_ne  = (const float*)d_in[24];
  const float* b_ne  = (const float*)d_in[25];
  const float* W_ea  = (const float*)d_in[26];
  const float* b_ea  = (const float*)d_in[27];
  const float* Wi_n  = (const float*)d_in[28];
  const float* Wh_n  = (const float*)d_in[29];
  const float* b_n   = (const float*)d_in[30];
  const float* W_out = (const float*)d_in[31];
  const float* b_out = (const float*)d_in[32];
  const int* srcp  = (const int*)d_in[33];
  const int* dstp  = (const int*)d_in[34];
  const int* intra = (const int*)d_in[35];
  const int* inter = (const int*)d_in[36];

  unsigned* ws = (unsigned*)d_ws;
  float* entp   = (float*)d_ws;       // first 128 floats
  unsigned* WT  = ws + 256;
  unsigned* WS  = WT  + 49152;        // 192x512
  unsigned* WN  = WS  + 49152;        // 192x512
  unsigned* WQ  = WN  + 65536;        // 256x512
  unsigned* WKI = WQ  + 4096;         // 128x64
  unsigned* WKE = WKI + 4096;
  unsigned* WEA = WKE + 4096;
  unsigned* WSE = WEA + 12288;        // 384x64
  float* cs = (float*)(WSE + 1024);   // 32x64 ; cs = 128*144*128 floats

  auto packs = [&](const float* Wi, const float* Wh, int Kin, int Kreal, int Ktot, int N, unsigned* dstw) {
    int total = (Ktot / 32) * (N / 16) * 256;
    pack_kernel<<<(total + 255) / 256, 256, 0, stream>>>(Wi, Wh, Kin, Kreal, Ktot, N, dstw);
  };
  packs(Wi_t, Wh_t, 64, 192, 192, 512, WT);
  packs(Wi_s, Wh_s, 64, 192, 192, 512, WS);
  packs(Wi_n, Wh_n, 128, 256, 256, 512, WN);
  packs(W_q,  nullptr, 128, 128, 128, 64, WQ);
  packs(W_ki, nullptr, 128, 128, 128, 64, WKI);
  packs(W_ke, nullptr, 128, 128, 128, 64, WKE);
  packs(W_ea, nullptr, 384, 384, 384, 64, WEA);
  packs(W_se, nullptr, 19, 19, 32, 64, WSE);

  P p;
  p.nodes = nodes; p.et = et; p.es = es;
  p.kp_emb = kp_emb; p.W_te = W_te; p.b_te = b_te; p.b_t = b_t;
  p.b_se = b_se; p.b_s = b_s;
  p.b_q = b_q; p.b_ki = b_ki; p.b_ke = b_ke; p.v_i = v_i; p.v_e = v_e;
  p.W_ne = W_ne; p.b_ne = b_ne; p.b_ea = b_ea; p.b_n = b_n;
  p.W_out = W_out; p.b_out = b_out;
  p.src = srcp; p.dst = dstp; p.intra = intra; p.inter = inter;
  p.WT = WT; p.WS = WS; p.WN = WN; p.WQ = WQ; p.WKI = WKI; p.WKE = WKE; p.WEA = WEA; p.WSE = WSE;
  p.cs = cs; p.entp = entp; p.out = (float*)d_out;

  srnn_kernel<<<128, NTHREADS, 0, stream>>>(p);
  fin_kernel<<<1, 64, 0, stream>>>(entp, (float*)d_out);
}
